// SelfAttention_22162031248051
// MI455X (gfx1250) — matrix operand order repaired and checked by reading, not compiled, not run on hardware
//
#include <hip/hip_runtime.h>

typedef __bf16 bf16_t;
typedef __attribute__((ext_vector_type(16))) __bf16 v16bf;
typedef __attribute__((ext_vector_type(8)))  __bf16 v8bf;
typedef __attribute__((ext_vector_type(8)))  float  v8f;
typedef __attribute__((ext_vector_type(4)))  float  v4f;
typedef __attribute__((ext_vector_type(4)))  unsigned int u32x4;
typedef __attribute__((ext_vector_type(8)))  int    i32x8;
typedef __attribute__((ext_vector_type(4)))  int    i32x4;

#define B_  4
#define S_  2048
#define D_  1024
#define H_  16
#define HD_ 64

#if defined(__has_builtin)
#if __has_builtin(__builtin_amdgcn_tensor_load_to_lds) && __has_builtin(__builtin_amdgcn_s_wait_tensorcnt)
#define USE_TDM 1
#endif
#endif

// ---------------- WMMA fragment loaders (wave32, 16x16x32 bf16) ----------------
// A matrix 16x32 (MxK), CDNA5 ISA layout:
//   lane<16 : M=lane,    elems[0:8)=K[k0..k0+7],      elems[8:16)=K[k0+16..k0+23]
//   lane>=16: M=lane-16, elems[0:8)=K[k0+8..k0+15],   elems[8:16)=K[k0+24..k0+31]
static __device__ inline v16bf load_a_bf16(const bf16_t* base, int ldk, int row0,
                                           int k0, int lane) {
  const int half = lane >> 4, m = lane & 15;
  const bf16_t* p = base + (size_t)(row0 + m) * ldk + k0 + half * 8;
  union { v16bf v; v8bf h[2]; } u;
  u.h[0] = *(const v8bf*)(p);
  u.h[1] = *(const v8bf*)(p + 16);
  return u.v;
}

static __device__ inline v16bf load_a_f32(const float* base, int ldk, int row0,
                                          int k0, int lane) {
  const int half = lane >> 4, m = lane & 15;
  const float* p = base + (size_t)(row0 + m) * ldk + k0 + half * 8;
  v4f a0 = *(const v4f*)(p);
  v4f a1 = *(const v4f*)(p + 4);
  v4f b0 = *(const v4f*)(p + 16);
  v4f b1 = *(const v4f*)(p + 20);
  v16bf r;
#pragma unroll
  for (int i = 0; i < 4; ++i) {
    r[i]      = (__bf16)a0[i];
    r[4 + i]  = (__bf16)a1[i];
    r[8 + i]  = (__bf16)b0[i];
    r[12 + i] = (__bf16)b1[i];
  }
  return r;
}

// B matrix 32x16 (KxN) sourced from a K-contiguous [N][K] buffer:
//   lane<16 : N=lane,    elems[0:8)=K[k0..k0+7],      elems[8:16)=K[k0+16..k0+23]
//   lane>=16: N=lane-16, elems[0:8)=K[k0+8..k0+15],   elems[8:16)=K[k0+24..k0+31]
static __device__ inline v16bf load_b_bf16(const bf16_t* base, int ldk, int n0,
                                           int k0, int lane) {
  const int half = lane >> 4, n = lane & 15;
  const bf16_t* p = base + (size_t)(n0 + n) * ldk + k0 + half * 8;
  union { v16bf v; v8bf h[2]; } u;
  u.h[0] = *(const v8bf*)(p);
  u.h[1] = *(const v8bf*)(p + 16);
  return u.v;
}

static __device__ inline v8f wmma_bf16(v16bf a, v16bf b, v8f c) {
  return __builtin_amdgcn_wmma_f32_16x16x32_bf16(false, a, false, b,
                                                 (short)0, c, false, false);
}

// ------------- TDM staging of a [128 rows x 32 cols] bf16 tile into LDS -------------
// gsrc points at the tile start inside a row-major [rows_total][ld] bf16 tensor.
// Descriptor per CDNA5 ISA (2D tile, data_size=2B).
static __device__ inline void stage_w_tile(const bf16_t* gsrc, bf16_t* lds,
                                           int rows_total, int ld,
                                           int tid, int wave) {
#ifdef USE_TDM
  if (wave == 0) {
    unsigned long long ga = (unsigned long long)(__SIZE_TYPE__)gsrc;
    unsigned lad = (unsigned)(__SIZE_TYPE__)lds;   // low 32 bits of LDS aperture = LDS byte offset
    u32x4 g0;
    g0.x = 1u;                                     // count=1 valid descriptor
    g0.y = lad;                                    // lds_addr [63:32]
    g0.z = (unsigned)(ga & 0xffffffffu);           // global_addr [95:64]
    g0.w = (unsigned)((ga >> 32) & 0x01ffffffu) | 0x80000000u;  // addr[56:32] | type=2
    i32x8 g1;
    g1[0] = 0x00010000;                            // wg_mask=0, data_size=1 (2 bytes)
    g1[1] = (ld & 0xffff) << 16;                   // tensor_dim0[15:0] @ bits 63:48
    g1[2] = (rows_total & 0xffff) << 16;           // dim0[31:16]=0 | tensor_dim1[15:0]
    g1[3] = 32 << 16;                              // tensor_dim1[31:16]=0 | tile_dim0=32
    g1[4] = 128;                                   // tile_dim1=128, tile_dim2=0
    g1[5] = ld;                                    // tensor_dim0_stride[31:0]
    g1[6] = 0;                                     // stride hi, tensor_dim1_stride lo
    g1[7] = 0;
    i32x4 gz4 = {0, 0, 0, 0};
    i32x8 gz8 = {0, 0, 0, 0, 0, 0, 0, 0};
    __builtin_amdgcn_tensor_load_to_lds(g0, g1, gz4, gz4, gz8, 0);
  }
  __builtin_amdgcn_s_wait_tensorcnt((short)0);
#else
  // Fallback: cooperative vector copy (256 threads x 2 x 16B = 8KB)
  for (int i = tid; i < 128 * 32 / 8; i += 256) {
    int n = i >> 2, kk = (i & 3) * 8;
    *(v8bf*)(lds + n * 32 + kk) = *(const v8bf*)(gsrc + (size_t)n * ld + kk);
  }
#endif
}

// ---------------- weight convert + transpose: W[rows][cols] -> Wt[cols][rows] bf16 ----
__global__ void wt_convert_kernel(const float* __restrict__ W, bf16_t* __restrict__ Wt,
                                  int rows, int cols) {
  int idx = blockIdx.x * blockDim.x + threadIdx.x;
  if (idx >= rows * cols) return;
  int r = idx / cols, c = idx - r * cols;
  Wt[(size_t)c * rows + r] = (__bf16)W[idx];
}

// ---------------- QKV GEMM: [8192 x 1024] x [1024 x 3072] + bias -----------------
// 8 waves: 4 along M x 2 along N; wave tile 32x64 -> block tile 128x128.
// Weight tile (shared by the 4 M-waves) staged via TDM into LDS each K-step.
__global__ __launch_bounds__(256)
void qkv_gemm_kernel(const float* __restrict__ X, const bf16_t* __restrict__ Wt,
                     const float* __restrict__ bias,
                     bf16_t* __restrict__ Qb, bf16_t* __restrict__ Kb,
                     bf16_t* __restrict__ Vt) {
  __shared__ __align__(16) bf16_t ldsW[128 * 32];   // [n_local][k_local] tile

  const int tid  = threadIdx.x;
  const int lane = tid & 31;
  const int wave = tid >> 5;
  const int mw = wave & 3, nw = wave >> 2;
  const int Mb = blockIdx.x * 128 + mw * 32;
  const int Nblk = blockIdx.y * 128;

  v8f acc[2][4] = {};
  for (int k0 = 0; k0 < D_; k0 += 32) {
    __syncthreads();  // previous tile fully consumed before overwrite
    stage_w_tile(Wt + (size_t)Nblk * D_ + k0, ldsW, 3 * D_, D_, tid, wave);
    __syncthreads();  // tile visible to all waves

    if (k0 + 32 < D_)  // prefetch next A rows -> global_prefetch_b8
      __builtin_prefetch(X + (size_t)(Mb + (lane & 15)) * D_ + k0 + 32, 0, 0);

    v16bf a[2], b[4];
#pragma unroll
    for (int mt = 0; mt < 2; ++mt) a[mt] = load_a_f32(X, D_, Mb + mt * 16, k0, lane);
#pragma unroll
    for (int nt = 0; nt < 4; ++nt) b[nt] = load_b_bf16(ldsW, 32, nw * 64 + nt * 16, 0, lane);
#pragma unroll
    for (int mt = 0; mt < 2; ++mt)
#pragma unroll
      for (int nt = 0; nt < 4; ++nt)
        acc[mt][nt] = wmma_bf16(a[mt], b[nt], acc[mt][nt]);
  }

  const int half = lane >> 4, ln = lane & 15;
  const int Nb = Nblk + nw * 64;
#pragma unroll
  for (int mt = 0; mt < 2; ++mt) {
#pragma unroll
    for (int nt = 0; nt < 4; ++nt) {
#pragma unroll
      for (int r = 0; r < 8; ++r) {
        int grow = Mb + mt * 16 + r + 8 * half;       // 0..8191 (b*S+s)
        int gcol = Nb + nt * 16 + ln;                 // 0..3071
        float v = acc[mt][nt][r] + bias[gcol];
        bf16_t bv = (__bf16)v;
        int b = grow >> 11, s = grow & (S_ - 1);
        if (gcol < D_) {
          int h = gcol >> 6, hd = gcol & 63;
          Qb[(((size_t)(b * H_ + h)) * S_ + s) * HD_ + hd] = bv;
        } else if (gcol < 2 * D_) {
          int e = gcol - D_;
          int h = e >> 6, hd = e & 63;
          Kb[(((size_t)(b * H_ + h)) * S_ + s) * HD_ + hd] = bv;
        } else {
          int e = gcol - 2 * D_;
          int h = e >> 6, hd = e & 63;
          Vt[(((size_t)(b * H_ + h)) * HD_ + hd) * S_ + s] = bv;
        }
      }
    }
  }
}

// ---------------- Flash attention: one wave = 16 q rows, online softmax ----------
__global__ __launch_bounds__(256)
void attention_kernel(const bf16_t* __restrict__ Qb, const bf16_t* __restrict__ Kb,
                      const bf16_t* __restrict__ Vt, const int* __restrict__ use_mask_p,
                      bf16_t* __restrict__ AO) {
  __shared__ bf16_t lds_p[8 * 16 * 32];   // per-wave 16x32 P tile (C-layout -> A-layout)

  const int lane = threadIdx.x & 31;
  const int wave = threadIdx.x >> 5;
  const int half = lane >> 4, ln = lane & 15;
  const int bh = blockIdx.y;              // b*H + h
  const int b = bh >> 4, h = bh & 15;
  const int q0 = blockIdx.x * 128 + wave * 16;
  const int causal = use_mask_p[0];

  const bf16_t* Qh = Qb + (size_t)bh * S_ * HD_;
  const bf16_t* Kh = Kb + (size_t)bh * S_ * HD_;
  const bf16_t* Vh = Vt + (size_t)bh * HD_ * S_;
  bf16_t* pbuf = lds_p + wave * (16 * 32);

  v16bf aq[2];
#pragma unroll
  for (int c = 0; c < 2; ++c) aq[c] = load_a_bf16(Qh, HD_, q0, c * 32, lane);

  v8f o[4] = {};
  float m_r[8], l_r[8];
#pragma unroll
  for (int r = 0; r < 8; ++r) { m_r[r] = -3.0e38f; l_r[r] = 0.0f; }

  const float scale = 0.125f;  // 1/sqrt(64)
  const int kv_end = causal ? (q0 + 16) : S_;

  for (int kv0 = 0; kv0 < kv_end; kv0 += 32) {
    if (kv0 + 32 < kv_end) {   // prefetch next KV tiles
      __builtin_prefetch(Kh + (size_t)(kv0 + 32 + ln) * HD_, 0, 0);
      __builtin_prefetch(Vh + (size_t)ln * S_ + kv0 + 32, 0, 0);
    }
    v8f s0 = {}, s1 = {};
#pragma unroll
    for (int c = 0; c < 2; ++c) {
      v16bf bk0 = load_b_bf16(Kh, HD_, kv0,      c * 32, lane);
      v16bf bk1 = load_b_bf16(Kh, HD_, kv0 + 16, c * 32, lane);
      s0 = wmma_bf16(aq[c], bk0, s0);
      s1 = wmma_bf16(aq[c], bk1, s1);
    }

    float p0[8], p1[8];
#pragma unroll
    for (int r = 0; r < 8; ++r) {
      int qpos = q0 + r + 8 * half;
      float v0 = s0[r] * scale;
      float v1 = s1[r] * scale;
      if (causal && (kv0 + ln      > qpos)) v0 = -3.0e38f;
      if (causal && (kv0 + 16 + ln > qpos)) v1 = -3.0e38f;
      float mx = fmaxf(v0, v1);
      mx = fmaxf(mx, __shfl_xor(mx, 1, 32));
      mx = fmaxf(mx, __shfl_xor(mx, 2, 32));
      mx = fmaxf(mx, __shfl_xor(mx, 4, 32));
      mx = fmaxf(mx, __shfl_xor(mx, 8, 32));
      float mnew = fmaxf(m_r[r], mx);
      float e0 = __expf(v0 - mnew);
      float e1 = __expf(v1 - mnew);
      float rs = e0 + e1;
      rs += __shfl_xor(rs, 1, 32);
      rs += __shfl_xor(rs, 2, 32);
      rs += __shfl_xor(rs, 4, 32);
      rs += __shfl_xor(rs, 8, 32);
      float f = __expf(m_r[r] - mnew);
      l_r[r] = l_r[r] * f + rs;
      m_r[r] = mnew;
      p0[r] = e0; p1[r] = e1;
#pragma unroll
      for (int nt = 0; nt < 4; ++nt) o[nt][r] *= f;
    }

    // C-layout -> LDS (row-major 16x32), then re-read as A fragment (per-wave, DS in-order)
#pragma unroll
    for (int r = 0; r < 8; ++r) {
      int row = r + 8 * half;
      pbuf[row * 32 + ln]      = (__bf16)p0[r];
      pbuf[row * 32 + 16 + ln] = (__bf16)p1[r];
    }
    v16bf ap = load_a_bf16(pbuf, 32, 0, 0, lane);

#pragma unroll
    for (int nt = 0; nt < 4; ++nt) {
      v16bf bv = load_b_bf16(Vh, S_, nt * 16, kv0, lane);
      o[nt] = wmma_bf16(ap, bv, o[nt]);
    }
  }

#pragma unroll
  for (int r = 0; r < 8; ++r) {
    float inv = 1.0f / l_r[r];
    int qpos = q0 + r + 8 * half;
#pragma unroll
    for (int nt = 0; nt < 4; ++nt) {
      int hd = nt * 16 + ln;
      AO[((size_t)(b * S_ + qpos)) * D_ + h * HD_ + hd] = (__bf16)(o[nt][r] * inv);
    }
  }
}

// ---------------- Output projection: [8192 x 1024] x [1024 x 1024] + bias --------
__global__ __launch_bounds__(256)
void out_proj_kernel(const bf16_t* __restrict__ A, const bf16_t* __restrict__ Wt,
                     const float* __restrict__ bias, float* __restrict__ out) {
  __shared__ __align__(16) bf16_t ldsW[128 * 32];

  const int tid  = threadIdx.x;
  const int lane = tid & 31;
  const int wave = tid >> 5;
  const int mw = wave & 3, nw = wave >> 2;
  const int Mb = blockIdx.x * 128 + mw * 32;
  const int Nblk = blockIdx.y * 128;

  v8f acc[2][4] = {};
  for (int k0 = 0; k0 < D_; k0 += 32) {
    __syncthreads();
    stage_w_tile(Wt + (size_t)Nblk * D_ + k0, ldsW, D_, D_, tid, wave);
    __syncthreads();

    if (k0 + 32 < D_)
      __builtin_prefetch(A + (size_t)(Mb + (lane & 15)) * D_ + k0 + 32, 0, 0);

    v16bf a[2], b[4];
#pragma unroll
    for (int mt = 0; mt < 2; ++mt) a[mt] = load_a_bf16(A, D_, Mb + mt * 16, k0, lane);
#pragma unroll
    for (int nt = 0; nt < 4; ++nt) b[nt] = load_b_bf16(ldsW, 32, nw * 64 + nt * 16, 0, lane);
#pragma unroll
    for (int mt = 0; mt < 2; ++mt)
#pragma unroll
      for (int nt = 0; nt < 4; ++nt)
        acc[mt][nt] = wmma_bf16(a[mt], b[nt], acc[mt][nt]);
  }

  const int half = lane >> 4, ln = lane & 15;
  const int Nb = Nblk + nw * 64;
#pragma unroll
  for (int mt = 0; mt < 2; ++mt)
#pragma unroll
    for (int nt = 0; nt < 4; ++nt)
#pragma unroll
      for (int r = 0; r < 8; ++r) {
        int grow = Mb + mt * 16 + r + 8 * half;
        int gcol = Nb + nt * 16 + ln;
        out[(size_t)grow * D_ + gcol] = acc[mt][nt][r] + bias[gcol];
      }
}

// ---------------------------------- launcher -------------------------------------
extern "C" void kernel_launch(void* const* d_in, const int* in_sizes, int n_in,
                              void* d_out, int out_size, void* d_ws, size_t ws_size,
                              hipStream_t stream) {
  const float* x     = (const float*)d_in[0];  // [B,S,D]
  const float* Wqkv  = (const float*)d_in[1];  // [D,3D]
  const float* bqkv  = (const float*)d_in[2];  // [3D]
  const float* Wout  = (const float*)d_in[3];  // [D,D]
  const float* bout  = (const float*)d_in[4];  // [D]
  const int*   msk   = (const int*)d_in[5];    // scalar
  float* out = (float*)d_out;

  bf16_t* WqkvT = (bf16_t*)d_ws;                               // [3D][D]
  bf16_t* WoutT = WqkvT + (size_t)3 * D_ * D_;                 // [D][D]
  bf16_t* Qb    = WoutT + (size_t)D_ * D_;                     // [B,H,S,HD]
  bf16_t* Kb    = Qb + (size_t)B_ * H_ * S_ * HD_;             // [B,H,S,HD]
  bf16_t* Vt    = Kb + (size_t)B_ * H_ * S_ * HD_;             // [B,H,HD,S]
  bf16_t* AO    = Vt + (size_t)B_ * H_ * S_ * HD_;             // [B,S,D]

  wt_convert_kernel<<<(D_ * 3 * D_ + 255) / 256, 256, 0, stream>>>(Wqkv, WqkvT, D_, 3 * D_);
  wt_convert_kernel<<<(D_ * D_ + 255) / 256, 256, 0, stream>>>(Wout, WoutT, D_, D_);

  qkv_gemm_kernel<<<dim3((B_ * S_) / 128, (3 * D_) / 128), 256, 0, stream>>>(
      x, WqkvT, bqkv, Qb, Kb, Vt);

  attention_kernel<<<dim3(S_ / 128, B_ * H_), 256, 0, stream>>>(Qb, Kb, Vt, msk, AO);

  out_proj_kernel<<<dim3((B_ * S_) / 128, D_ / 128), 256, 0, stream>>>(AO, WoutT, bout, out);
}
